// MelDecoder_81441169866910
// MI455X (gfx1250) — matrix operand order repaired and checked by reading, not compiled, not run on hardware
//
#include <hip/hip_runtime.h>
#include <hip/hip_bf16.h>

// ---------------------------------------------------------------------------
// MI455X (gfx1250, wave32) mel-decoder.
// bf16 WMMA (v_wmma_f32_16x16x32_bf16) for all GEMMs. All GEMM A-operands are
// materialized in bf16 by their producer stage so the K-loop is pure
// global_load_b128 -> v_wmma (no per-element f32->bf16 VALU in the hot loop).
// keys+memory live in L2 as bf16 (65 MB < 192 MB L2): the 200-step attention
// scan is L2-resident streaming, not HBM bound.
// ---------------------------------------------------------------------------

typedef __attribute__((ext_vector_type(16))) __bf16 v16bf;
typedef __attribute__((ext_vector_type(8)))  float  v8f;

#define B_    64
#define T_    1000
#define U_    256
#define H_    128
#define MEL_  80
#define R_    5
#define STEPS_ (T_ / R_)       // 200
#define G3_   768              // 3*U
#define KP1_  96               // MEL=80 padded to multiple of 32 for WMMA K

__device__ __forceinline__ unsigned short f2bf(float f) {
  unsigned u = __float_as_uint(f);
  u += 0x7FFFu + ((u >> 16) & 1u);           // RNE truncation to bf16
  return (unsigned short)(u >> 16);
}
__device__ __forceinline__ float bf2f(unsigned short s) {
  return __uint_as_float(((unsigned)s) << 16);
}
__device__ __forceinline__ float sigm(float x) { return 1.f / (1.f + __expf(-x)); }

__device__ __forceinline__ void wg_sync() { __threadfence(); __syncthreads(); }

// --------------------------- WMMA fragment packing -------------------------
// 16-bit A (16x32, MxK), ISA 7.12.2: lanes 0-15 hold M=lane, K=0..7 & 16..23;
// lanes 16-31 hold M=lane-16, K=8..15 & 24..31.
__device__ __forceinline__ v16bf a_frag_u16(const unsigned short* A, int lda, int lane) {
  const int m = lane & 15, kb = (lane < 16) ? 0 : 8;
  union { v16bf v; unsigned short s[16]; } u;
#pragma unroll
  for (int j = 0; j < 8; ++j) {
    u.s[j]     = A[m * lda + kb + j];
    u.s[j + 8] = A[m * lda + kb + 16 + j];
  }
  return u.v;
}
// 16-bit B (32x16, KxN): lane n holds column N=n%16; K=0..7 & 16..23 (lanes 0-15)
// or K=8..15 & 24..31 (lanes 16-31), matching the A layout. Source W is [N,K]
// row-major (torch weight [out,in]).
__device__ __forceinline__ v16bf b_frag_u16(const unsigned short* W, int ldw, int lane) {
  const int n = lane & 15, kb = (lane < 16) ? 0 : 8;
  union { v16bf v; unsigned short s[16]; } u;
#pragma unroll
  for (int j = 0; j < 8; ++j) { u.s[j] = W[n * ldw + kb + j]; u.s[j + 8] = W[n * ldw + kb + 16 + j]; }
  return u.v;
}

// Wave-per-tile GEMM: A[M,K](bf16) * W[N,K]^T(bf16) + bias, optional relu.
// Stores f32 (Cf) and/or bf16 (Cb) outputs.
__device__ __forceinline__ void gemm_bf(const unsigned short* A, int lda,
                                        const unsigned short* Wb, int ldw,
                                        const float* bias,
                                        float* Cf, unsigned short* Cb, int ldc,
                                        int Mtiles, int Ntiles, int K, int relu,
                                        int wid, int nwaves, int lane) {
  const int ntot = Mtiles * Ntiles;
  for (int t = wid; t < ntot; t += nwaves) {
    const int mt = t % Mtiles, nt = t / Mtiles;
    const unsigned short* Ap = A + (size_t)mt * 16 * lda;
    const unsigned short* Wp = Wb + (size_t)nt * 16 * ldw;
    v8f acc = {};
    for (int k = 0; k < K; k += 32) {
      v16bf af = a_frag_u16(Ap + k, lda, lane);
      v16bf bf = b_frag_u16(Wp + k, ldw, lane);
      acc = __builtin_amdgcn_wmma_f32_16x16x32_bf16(false, af, false, bf,
                                                    (short)0, acc, false, false);
    }
    const int n = lane & 15, mb = (lane < 16) ? 0 : 8;
#pragma unroll
    for (int i = 0; i < 8; ++i) {
      float vv = acc[i] + (bias ? bias[nt * 16 + n] : 0.f);
      if (relu) vv = fmaxf(vv, 0.f);
      const size_t idx = (size_t)(mt * 16 + mb + i) * ldc + nt * 16 + n;
      if (Cf) Cf[idx] = vv;
      if (Cb) Cb[idx] = f2bf(vv);
    }
  }
}

// ------------------------------- utility kernels ---------------------------
__global__ void k_f2bf_pad(const float* __restrict__ src, unsigned short* __restrict__ dst,
                           int rows, int ksrc, int kpad) {
  int total = rows * kpad;
  for (int i = blockIdx.x * blockDim.x + threadIdx.x; i < total;
       i += gridDim.x * blockDim.x) {
    int r = i / kpad, k = i - r * kpad;
    dst[i] = (k < ksrc) ? f2bf(src[(size_t)r * ksrc + k]) : (unsigned short)0;
  }
}
__global__ void k_zero(float* p, int n) {
  for (int i = blockIdx.x * blockDim.x + threadIdx.x; i < n; i += gridDim.x * blockDim.x)
    p[i] = 0.f;
}
__global__ void k_zero_u16(unsigned short* p, int n) {
  for (int i = blockIdx.x * blockDim.x + threadIdx.x; i < n; i += gridDim.x * blockDim.x)
    p[i] = 0;
}

// keys = memory @ W1^T, output bf16.  grid = (M/16, N/16), one wave per block.
__global__ __launch_bounds__(32) void k_keys(const unsigned short* __restrict__ mem_bf,
                                             const unsigned short* __restrict__ w1b,
                                             unsigned short* __restrict__ keys_bf) {
  const int mt = blockIdx.x, nt = blockIdx.y, lane = threadIdx.x;
  const unsigned short* Ap = mem_bf + (size_t)mt * 16 * U_;
  const unsigned short* Wp = w1b + (size_t)nt * 16 * U_;
  v8f acc = {};
  for (int k = 0; k < U_; k += 32) {
    v16bf af = a_frag_u16(Ap + k, U_, lane);
    v16bf bf = b_frag_u16(Wp + k, U_, lane);
    acc = __builtin_amdgcn_wmma_f32_16x16x32_bf16(false, af, false, bf,
                                                  (short)0, acc, false, false);
  }
  const int n = lane & 15, mb = (lane < 16) ? 0 : 8;
#pragma unroll
  for (int i = 0; i < 8; ++i)
    keys_bf[((size_t)mt * 16 + mb + i) * U_ + nt * 16 + n] = f2bf(acc[i]);
}

// Per-step: prenet + attention-GRU + q   (single 32-wave workgroup)
__global__ __launch_bounds__(1024) void k_stepA(
    const unsigned short* __restrict__ prev_bf,
    unsigned short* x1b, unsigned short* x2b,
    float* attn_h, unsigned short* attn_hb, float* q,
    float* gi, float* gh,
    const unsigned short* Wp1b, const unsigned short* Wp2b,
    const unsigned short* Waihb, const unsigned short* Wahhb, const unsigned short* W2b,
    const float* bp1, const float* bp2, const float* ba_ih, const float* ba_hh) {
  const int tid = threadIdx.x, lane = tid & 31, wid = tid >> 5;
  const int NW = 32;
  // x1 = relu(prev @ Wp1^T + bp1), prev padded [64,96]; bf16-only output
  gemm_bf(prev_bf, KP1_, Wp1b, KP1_, bp1, nullptr, x1b, U_,
          B_ / 16, U_ / 16, KP1_, 1, wid, NW, lane);
  wg_sync();
  // x2 = relu(x1 @ Wp2^T + bp2); bf16-only output
  gemm_bf(x1b, U_, Wp2b, U_, bp2, nullptr, x2b, H_,
          B_ / 16, H_ / 16, U_, 1, wid, NW, lane);
  wg_sync();
  // gates of attention GRU (f32 outputs for the nonlinearity)
  gemm_bf(x2b, H_, Waihb, H_, ba_ih, gi, nullptr, G3_,
          B_ / 16, G3_ / 16, H_, 0, wid, NW, lane);
  gemm_bf(attn_hb, U_, Wahhb, U_, ba_hh, gh, nullptr, G3_,
          B_ / 16, G3_ / 16, U_, 0, wid, NW, lane);
  wg_sync();
  for (int i = tid; i < B_ * U_; i += 1024) {
    int b = i / U_, u = i - b * U_;
    float r = sigm(gi[b * G3_ + u] + gh[b * G3_ + u]);
    float z = sigm(gi[b * G3_ + U_ + u] + gh[b * G3_ + U_ + u]);
    float n = tanhf(gi[b * G3_ + 2 * U_ + u] + r * gh[b * G3_ + 2 * U_ + u]);
    float h = attn_h[i];
    float hn = (1.f - z) * n + z * h;      // d_t == new attn hidden
    attn_h[i] = hn;
    attn_hb[i] = f2bf(hn);
  }
  wg_sync();
  // q = d_t @ W2^T
  gemm_bf(attn_hb, U_, W2b, U_, nullptr, q, nullptr, U_,
          B_ / 16, U_ / 16, U_, 0, wid, NW, lane);
}

// Per-step attention: one block per batch row. scores->softmax->context.
__global__ __launch_bounds__(256) void k_attn(const unsigned short* __restrict__ keys_bf,
                                              const unsigned short* __restrict__ mem_bf,
                                              const float* __restrict__ q,
                                              const float* __restrict__ vw,
                                              float* __restrict__ ctx) {
  const int b = blockIdx.x, tid = threadIdx.x;
  const int lane = tid & 31, wid = tid >> 5;
  __shared__ float q_s[U_], v_s[U_], sc[T_], red[8];
  q_s[tid] = q[b * U_ + tid];
  v_s[tid] = vw[tid];
  __syncthreads();

  float lmax = -3.0e38f;
  for (int t = tid; t < T_; t += 256) {
    const unsigned short* kp = keys_bf + ((size_t)b * T_ + t) * U_;
    __builtin_prefetch(kp + 256 * U_, 0, 0);   // global_prefetch of next keys row
    float s = 0.f;
#pragma unroll 4
    for (int u = 0; u < U_; u += 2) {
      unsigned pk = *(const unsigned*)(kp + u);
      float k0 = bf2f((unsigned short)(pk & 0xffffu));
      float k1 = bf2f((unsigned short)(pk >> 16));
      s += tanhf(k0 + q_s[u]) * v_s[u];
      s += tanhf(k1 + q_s[u + 1]) * v_s[u + 1];
    }
    sc[t] = s;
    lmax = fmaxf(lmax, s);
  }
  for (int o = 16; o > 0; o >>= 1) lmax = fmaxf(lmax, __shfl_xor(lmax, o, 32));
  if (lane == 0) red[wid] = lmax;
  __syncthreads();
  float gmax = red[0];
#pragma unroll
  for (int i = 1; i < 8; ++i) gmax = fmaxf(gmax, red[i]);
  __syncthreads();

  float lsum = 0.f;
  for (int t = tid; t < T_; t += 256) {
    float e = __expf(sc[t] - gmax);
    sc[t] = e;
    lsum += e;
  }
  for (int o = 16; o > 0; o >>= 1) lsum += __shfl_xor(lsum, o, 32);
  if (lane == 0) red[wid] = lsum;
  __syncthreads();
  float gsum = 0.f;
#pragma unroll
  for (int i = 0; i < 8; ++i) gsum += red[i];
  const float inv = 1.f / gsum;

  // ctx[u] = (1/sum) * sum_t e[t] * memory[b,t,u]; u = tid, coalesced over lanes
  const unsigned short* mp = mem_bf + (size_t)b * T_ * U_ + tid;
  float acc = 0.f;
  for (int t = 0; t < T_; ++t) acc += sc[t] * bf2f(mp[(size_t)t * U_]);
  ctx[b * U_ + tid] = acc * inv;
}

// Per-step: proj + GRU1 + GRU2 + output + scatter to d_out, update prev frame.
__global__ __launch_bounds__(1024) void k_stepC(
    const float* __restrict__ attn_h, const float* __restrict__ ctx,
    unsigned short* catb,
    float* g1_in, unsigned short* g1_inb, float* g1_h, unsigned short* g1_hb,
    float* gi1, float* gh1,
    float* g2_in, unsigned short* g2_inb, float* g2_h, unsigned short* g2_hb,
    float* gi2, float* gh2, unsigned short* ybb,
    unsigned short* prev_bf, float* __restrict__ out,
    const unsigned short* Wprojb, const unsigned short* W1ihb, const unsigned short* W1hhb,
    const unsigned short* W2ihb, const unsigned short* W2hhb, const unsigned short* Woutb,
    const float* bproj, const float* b1_ih, const float* b1_hh,
    const float* b2_ih, const float* b2_hh, const float* bout, int step) {
  const int tid = threadIdx.x, lane = tid & 31, wid = tid >> 5;
  const int NW = 32;
  // cat = [d_t | ctx] in bf16
  for (int i = tid; i < B_ * 2 * U_; i += 1024) {
    int b = i / (2 * U_), k = i - b * 2 * U_;
    catb[i] = f2bf((k < U_) ? attn_h[b * U_ + k] : ctx[b * U_ + k - U_]);
  }
  wg_sync();
  gemm_bf(catb, 2 * U_, Wprojb, 2 * U_, bproj, g1_in, g1_inb, U_,
          4, 16, 2 * U_, 0, wid, NW, lane);
  wg_sync();
  gemm_bf(g1_inb, U_, W1ihb, U_, b1_ih, gi1, nullptr, G3_, 4, 48, U_, 0, wid, NW, lane);
  gemm_bf(g1_hb, U_, W1hhb, U_, b1_hh, gh1, nullptr, G3_, 4, 48, U_, 0, wid, NW, lane);
  wg_sync();
  for (int i = tid; i < B_ * U_; i += 1024) {
    int b = i / U_, u = i - b * U_;
    float r = sigm(gi1[b * G3_ + u] + gh1[b * G3_ + u]);
    float z = sigm(gi1[b * G3_ + U_ + u] + gh1[b * G3_ + U_ + u]);
    float n = tanhf(gi1[b * G3_ + 2 * U_ + u] + r * gh1[b * G3_ + 2 * U_ + u]);
    float hn = (1.f - z) * n + z * g1_h[i];
    g1_h[i] = hn;
    g1_hb[i] = f2bf(hn);
    float gg = g1_in[i] + hn;
    g2_in[i] = gg;
    g2_inb[i] = f2bf(gg);
  }
  wg_sync();
  gemm_bf(g2_inb, U_, W2ihb, U_, b2_ih, gi2, nullptr, G3_, 4, 48, U_, 0, wid, NW, lane);
  gemm_bf(g2_hb, U_, W2hhb, U_, b2_hh, gh2, nullptr, G3_, 4, 48, U_, 0, wid, NW, lane);
  wg_sync();
  for (int i = tid; i < B_ * U_; i += 1024) {
    int b = i / U_, u = i - b * U_;
    float r = sigm(gi2[b * G3_ + u] + gh2[b * G3_ + u]);
    float z = sigm(gi2[b * G3_ + U_ + u] + gh2[b * G3_ + U_ + u]);
    float n = tanhf(gi2[b * G3_ + 2 * U_ + u] + r * gh2[b * G3_ + 2 * U_ + u]);
    float hn = (1.f - z) * n + z * g2_h[i];
    g2_h[i] = hn;
    g2_hb[i] = f2bf(hn);
    ybb[i] = f2bf(g2_in[i] + hn);
  }
  wg_sync();
  // out = yb @ Wout^T + bout -> [64,400]; scatter col = mel*R + r into d_out
  for (int t = wid; t < 4 * 25; t += NW) {
    int mt = t % 4, nt = t / 4;
    const unsigned short* Ap = ybb + (size_t)mt * 16 * U_;
    const unsigned short* Wp = Woutb + (size_t)nt * 16 * U_;
    v8f acc = {};
    for (int k = 0; k < U_; k += 32) {
      v16bf af = a_frag_u16(Ap + k, U_, lane);
      v16bf bf = b_frag_u16(Wp + k, U_, lane);
      acc = __builtin_amdgcn_wmma_f32_16x16x32_bf16(false, af, false, bf,
                                                    (short)0, acc, false, false);
    }
    const int n = lane & 15, mb = (lane < 16) ? 0 : 8;
#pragma unroll
    for (int i = 0; i < 8; ++i) {
      int b = mt * 16 + mb + i, col = nt * 16 + n;
      float vv = acc[i] + bout[col];
      int mel = col / R_, r = col - mel * R_;
      out[(size_t)b * (MEL_ * T_) + (size_t)mel * T_ + step * R_ + r] = vv;
      if (r == R_ - 1) prev_bf[b * KP1_ + mel] = f2bf(vv);  // next GO frame
    }
  }
}

// ------------------------------------ host ---------------------------------
extern "C" void kernel_launch(void* const* d_in, const int* in_sizes, int n_in,
                              void* d_out, int out_size, void* d_ws, size_t ws_size,
                              hipStream_t stream) {
  (void)in_sizes; (void)n_in; (void)out_size; (void)ws_size;
  const float* memory = (const float*)d_in[0];
  const float* Wp1 = (const float*)d_in[1];  const float* bp1 = (const float*)d_in[2];
  const float* Wp2 = (const float*)d_in[3];  const float* bp2 = (const float*)d_in[4];
  const float* W1  = (const float*)d_in[5];  const float* W2  = (const float*)d_in[6];
  const float* vv  = (const float*)d_in[7];
  const float* Wa_ih = (const float*)d_in[8];  const float* Wa_hh = (const float*)d_in[9];
  const float* ba_ih = (const float*)d_in[10]; const float* ba_hh = (const float*)d_in[11];
  const float* W1_ih = (const float*)d_in[12]; const float* W1_hh = (const float*)d_in[13];
  const float* b1_ih = (const float*)d_in[14]; const float* b1_hh = (const float*)d_in[15];
  const float* W2_ih = (const float*)d_in[16]; const float* W2_hh = (const float*)d_in[17];
  const float* b2_ih = (const float*)d_in[18]; const float* b2_hh = (const float*)d_in[19];
  const float* Wproj = (const float*)d_in[20]; const float* bproj = (const float*)d_in[21];
  const float* Wout  = (const float*)d_in[22]; const float* bout  = (const float*)d_in[23];

  char* base = (char*)d_ws;
  size_t off = 0;
  auto a_u16 = [&](size_t n) { unsigned short* p = (unsigned short*)(base + off);
                               off += ((n * 2 + 255) & ~(size_t)255); return p; };
  auto a_f32 = [&](size_t n) { float* p = (float*)(base + off);
                               off += ((n * 4 + 255) & ~(size_t)255); return p; };

  unsigned short* mem_bf  = a_u16((size_t)B_ * T_ * U_);
  unsigned short* keys_bf = a_u16((size_t)B_ * T_ * U_);
  unsigned short* W1b    = a_u16(U_ * U_);
  unsigned short* Wp1b   = a_u16(U_ * KP1_);
  unsigned short* Wp2b   = a_u16(H_ * U_);
  unsigned short* W2b    = a_u16(U_ * U_);
  unsigned short* Waihb  = a_u16(G3_ * H_);
  unsigned short* Wahhb  = a_u16(G3_ * U_);
  unsigned short* W1ihb  = a_u16(G3_ * U_);
  unsigned short* W1hhb  = a_u16(G3_ * U_);
  unsigned short* W2ihb  = a_u16(G3_ * U_);
  unsigned short* W2hhb  = a_u16(G3_ * U_);
  unsigned short* Wprojb = a_u16(U_ * 2 * U_);
  unsigned short* Woutb  = a_u16(MEL_ * R_ * U_);

  // bf16 activation/state buffers (GEMM A operands)
  unsigned short* prev_bf = a_u16(B_ * KP1_);
  unsigned short* x1b     = a_u16(B_ * U_);
  unsigned short* x2b     = a_u16(B_ * H_);
  unsigned short* attn_hb = a_u16(B_ * U_);
  unsigned short* catb    = a_u16(B_ * 2 * U_);
  unsigned short* g1_inb  = a_u16(B_ * U_);
  unsigned short* g1_hb   = a_u16(B_ * U_);
  unsigned short* g2_inb  = a_u16(B_ * U_);
  unsigned short* g2_hb   = a_u16(B_ * U_);
  unsigned short* ybb     = a_u16(B_ * U_);

  // f32 buffers (nonlinearity inputs + recurrent state kept in f32)
  float* attn_h = a_f32(B_ * U_);
  float* g1_h   = a_f32(B_ * U_);
  float* g2_h   = a_f32(B_ * U_);
  float* gi_a   = a_f32(B_ * G3_);
  float* gh_a   = a_f32(B_ * G3_);
  float* q      = a_f32(B_ * U_);
  float* ctx    = a_f32(B_ * U_);
  float* g1_in  = a_f32(B_ * U_);
  float* gi1    = a_f32(B_ * G3_);
  float* gh1    = a_f32(B_ * G3_);
  float* g2_in  = a_f32(B_ * U_);
  float* gi2    = a_f32(B_ * G3_);
  float* gh2    = a_f32(B_ * G3_);

  auto cv = [&](const float* src, unsigned short* dst, int rows, int ksrc, int kpad) {
    int total = rows * kpad;
    int blocks = (total + 255) / 256; if (blocks > 4096) blocks = 4096;
    k_f2bf_pad<<<blocks, 256, 0, stream>>>(src, dst, rows, ksrc, kpad);
  };
  cv(memory, mem_bf, B_ * T_, U_, U_);
  cv(W1, W1b, U_, U_, U_);
  cv(Wp1, Wp1b, U_, MEL_, KP1_);
  cv(Wp2, Wp2b, H_, U_, U_);
  cv(W2, W2b, U_, U_, U_);
  cv(Wa_ih, Waihb, G3_, H_, H_);
  cv(Wa_hh, Wahhb, G3_, U_, U_);
  cv(W1_ih, W1ihb, G3_, U_, U_);
  cv(W1_hh, W1hhb, G3_, U_, U_);
  cv(W2_ih, W2ihb, G3_, U_, U_);
  cv(W2_hh, W2hhb, G3_, U_, U_);
  cv(Wproj, Wprojb, U_, 2 * U_, 2 * U_);
  cv(Wout, Woutb, MEL_ * R_, U_, U_);

  // zero recurrent state every launch (ws is poisoned once by harness)
  k_zero<<<(B_ * U_ * 3 + 255) / 256, 256, 0, stream>>>(attn_h, B_ * U_);
  k_zero<<<(B_ * U_ + 255) / 256, 256, 0, stream>>>(g1_h, B_ * U_);
  k_zero<<<(B_ * U_ + 255) / 256, 256, 0, stream>>>(g2_h, B_ * U_);
  k_zero_u16<<<(B_ * KP1_ + 255) / 256, 256, 0, stream>>>(prev_bf, B_ * KP1_);
  k_zero_u16<<<(B_ * U_ + 255) / 256, 256, 0, stream>>>(attn_hb, B_ * U_);
  k_zero_u16<<<(B_ * U_ + 255) / 256, 256, 0, stream>>>(g1_hb, B_ * U_);
  k_zero_u16<<<(B_ * U_ + 255) / 256, 256, 0, stream>>>(g2_hb, B_ * U_);

  // keys = memory @ W1^T  (hoisted out of the time loop)
  k_keys<<<dim3((B_ * T_) / 16, U_ / 16), 32, 0, stream>>>(mem_bf, W1b, keys_bf);

  for (int s = 0; s < STEPS_; ++s) {
    k_stepA<<<1, 1024, 0, stream>>>(prev_bf, x1b, x2b, attn_h, attn_hb, q, gi_a, gh_a,
                                    Wp1b, Wp2b, Waihb, Wahhb, W2b,
                                    bp1, bp2, ba_ih, ba_hh);
    k_attn<<<B_, 256, 0, stream>>>(keys_bf, mem_bf, q, vv, ctx);
    k_stepC<<<1, 1024, 0, stream>>>(attn_h, ctx, catb,
                                    g1_in, g1_inb, g1_h, g1_hb, gi1, gh1,
                                    g2_in, g2_inb, g2_h, g2_hb, gi2, gh2, ybb,
                                    prev_bf, (float*)d_out,
                                    Wprojb, W1ihb, W1hhb, W2ihb, W2hhb, Woutb,
                                    bproj, b1_ih, b1_hh, b2_ih, b2_hh, bout, s);
  }
}
